// EmformerAttention_74388833566790
// MI455X (gfx1250) — matrix operand order repaired and checked by reading, not compiled, not run on hardware
//
#include <hip/hip_runtime.h>
#include <hip/hip_bf16.h>
#include <stdint.h>

// ---------------------------------------------------------------------------
// Emformer attention for MI455X (gfx1250): bf16 WMMA GEMMs + flash attention
// D=1024 H=8 T=1024 R=32 S=8 M=8 B=16  -> Q = K = 1064, d = 128, B*H = 128
// ---------------------------------------------------------------------------

#define QTOT 1064
#define KTOT 1064
#define SEQP 1152          // padded sequence length (multiple of 128 and 64)
#define DHEAD 128
#define NHEADS 128         // B*H
#define ATT_SCALE 0.08838834764831845f
#define NEG_INF_F (-1.0e8f)
#define OUT_MEMS_BASE 17301504   // 1056*16*1024

typedef __attribute__((ext_vector_type(16))) __bf16 v16bf;
typedef __attribute__((ext_vector_type(8)))  float  v8f;

union Frag16 { v16bf v; uint32_t u[8]; };
union U4H8   { uint4 q; uint16_t h[8]; };

__device__ __forceinline__ uint16_t f2bf(float f) {
  uint32_t x = __float_as_uint(f);
  x += 0x7FFFu + ((x >> 16) & 1u);     // round-to-nearest-even
  return (uint16_t)(x >> 16);
}

// A-matrix 16x32 bf16 fragment: lane(0..15)=row M; VGPR j / lane-half -> K pair
__device__ __forceinline__ int a_koff(int j, int half) {
  return (j < 4) ? (2 * j + half * 8) : (16 + 2 * (j - 4) + half * 8);
}
// B-matrix 32x16 bf16 fragment: lane(0..15)=col N; VGPR j / lane-half -> K pair
__device__ __forceinline__ int b_koff(int j, int half) {
  return (j < 4) ? (2 * j + half * 8) : (16 + 2 * (j - 4) + half * 8);
}

// ---------------------------------------------------------------------------
// Elementwise helpers
// ---------------------------------------------------------------------------
__global__ void cvt_f32_bf16_kernel(const float* __restrict__ src,
                                    uint16_t* __restrict__ dst, int n) {
  int i = blockIdx.x * blockDim.x + threadIdx.x;
  int stride = gridDim.x * blockDim.x;
  for (; i < n; i += stride) dst[i] = f2bf(src[i]);
}

__global__ void zero_u32_kernel(uint32_t* __restrict__ p, int n) {
  int i = blockIdx.x * blockDim.x + threadIdx.x;
  int stride = gridDim.x * blockDim.x;
  for (; i < n; i += stride) p[i] = 0u;
}

// ---------------------------------------------------------------------------
// Concatenated-input row selectors (f32 sources)
// ---------------------------------------------------------------------------
__device__ __forceinline__ const float* qin_row(const float* u, const float* rc,
                                                const float* sm, int t, int b) {
  if (t < 32)   return rc + ((size_t)t * 16 + b) * 1024;
  if (t < 1056) return u  + ((size_t)(t - 32) * 16 + b) * 1024;
  return sm + ((size_t)(t - 1056) * 16 + b) * 1024;
}
__device__ __forceinline__ const float* kvin_row(const float* u, const float* rc,
                                                 const float* me, int t, int b) {
  if (t < 8)  return me + ((size_t)t * 16 + b) * 1024;
  if (t < 40) return rc + ((size_t)(t - 8) * 16 + b) * 1024;
  return u + ((size_t)(t - 40) * 16 + b) * 1024;
}

// ---------------------------------------------------------------------------
// Tiled bf16 WMMA GEMM: C[M=17024, N] = A[M,1024] * W[N,1024]^T + bias
// mode 0: A = concat(rc,utt,sum) f32 -> write q_heads bf16 [nh][t][e]
// mode 1: A = concat(mems,rc,utt) f32, N=2048 -> k_heads / v_heads bf16
// mode 2: A = attn bf16 -> f32 out with bias, clip, output/mems scatter
// Block: 256 threads (8 waves), tile 64(M) x 128(N), K staged 64 via LDS.
// ---------------------------------------------------------------------------
#define AS_STRIDE 72   // 144B row stride: 16B multiple, conflict-free (36i mod 64)
#define BS_STRIDE 72

__global__ __launch_bounds__(256)
void proj_gemm_kernel(int mode,
                      const float* __restrict__ utter,
                      const float* __restrict__ rctx,
                      const float* __restrict__ summ,
                      const float* __restrict__ mems,
                      const uint16_t* __restrict__ attn_bf,
                      const uint16_t* __restrict__ Wbf,
                      const float* __restrict__ bias,
                      uint16_t* __restrict__ q_h,
                      uint16_t* __restrict__ k_h,
                      uint16_t* __restrict__ v_h,
                      float* __restrict__ out_f32) {
  __shared__ __align__(16) uint16_t As[64 * AS_STRIDE];
  __shared__ __align__(16) uint16_t Bs[128 * BS_STRIDE];

  const int tid  = threadIdx.x;
  const int lane = tid & 31;
  const int w    = tid >> 5;
  const int half = lane >> 4;
  const int l15  = lane & 15;
  const int wm   = w & 1;           // 2 wave-rows  (x32 M)
  const int wn   = w >> 1;          // 4 wave-cols  (x32 N)
  const int M0   = blockIdx.x * 64;
  const int N0   = blockIdx.y * 128;

  // staging coordinates (fixed per thread)
  const int arow = tid >> 2, ac0 = (tid & 3) * 16;    // A: 64 rows x 64 k
  const int brow = tid >> 1, bc0 = (tid & 1) * 32;    // B: 128 rows x 64 k
  const int ar = M0 + arow, at = ar >> 4, ab = ar & 15;
  const float* asrcf = nullptr;
  if (mode == 0)      asrcf = qin_row(utter, rctx, summ, at, ab);
  else if (mode == 1) asrcf = kvin_row(utter, rctx, mems, at, ab);

  v8f acc[2][2];
  for (int a = 0; a < 2; a++)
    for (int b = 0; b < 2; b++)
      for (int e = 0; e < 8; e++) acc[a][b][e] = 0.0f;

  for (int k0 = 0; k0 < 1024; k0 += 64) {
    __syncthreads();
    // ---- stage A tile: 64 rows x 64 k  (convert f32->bf16 unless mode 2) ----
    {
      uint16_t* dst = &As[arow * AS_STRIDE + ac0];
      if (mode == 2) {
        const uint16_t* src = attn_bf + (size_t)ar * 1024 + k0 + ac0;
        *(uint4*)(dst)     = *(const uint4*)(src);
        *(uint4*)(dst + 8) = *(const uint4*)(src + 8);
      } else {
        const float* src = asrcf + k0 + ac0;
        for (int i = 0; i < 4; i++) {
          const float4 f = *(const float4*)(src + i * 4);
          uint2 p;
          p.x = (uint32_t)f2bf(f.x) | ((uint32_t)f2bf(f.y) << 16);
          p.y = (uint32_t)f2bf(f.z) | ((uint32_t)f2bf(f.w) << 16);
          *(uint2*)(dst + i * 4) = p;
        }
      }
    }
    // ---- stage B tile: 128 n-rows x 64 k (weights pre-converted to bf16) ----
    {
      const uint16_t* src = Wbf + (size_t)(N0 + brow) * 1024 + k0 + bc0;
      uint16_t* dst = &Bs[brow * BS_STRIDE + bc0];
      for (int i = 0; i < 4; i++)
        *(uint4*)(dst + i * 8) = *(const uint4*)(src + i * 8);
    }
    // prefetch next K-slab while this one is consumed
    if (k0 + 64 < 1024) {
      __builtin_prefetch(Wbf + (size_t)(N0 + brow) * 1024 + k0 + 64 + bc0, 0, 1);
      if (mode == 2)
        __builtin_prefetch(attn_bf + (size_t)ar * 1024 + k0 + 64 + ac0, 0, 1);
      else
        __builtin_prefetch(asrcf + k0 + 64 + ac0, 0, 1);
    }
    __syncthreads();

    for (int kc = 0; kc < 2; kc++) {
      Frag16 afr[2], bfr[2];
      for (int sm = 0; sm < 2; sm++) {
        const uint16_t* base = &As[(wm * 32 + sm * 16 + l15) * AS_STRIDE + kc * 32];
        for (int j = 0; j < 8; j++)
          afr[sm].u[j] = *(const uint32_t*)(base + a_koff(j, half));
      }
      for (int sn = 0; sn < 2; sn++) {
        const uint16_t* base = &Bs[(wn * 32 + sn * 16 + l15) * BS_STRIDE + kc * 32];
        for (int j = 0; j < 8; j++)
          bfr[sn].u[j] = *(const uint32_t*)(base + b_koff(j, half));
      }
      for (int sm = 0; sm < 2; sm++)
        for (int sn = 0; sn < 2; sn++)
          acc[sm][sn] = __builtin_amdgcn_wmma_f32_16x16x32_bf16(
              false, afr[sm].v, false, bfr[sn].v,
              (short)0, acc[sm][sn], false, false);
    }
  }

  // epilogue (C/D layout: VGPR i -> row i + 8*half, lane&15 -> col)
  for (int sm = 0; sm < 2; sm++)
    for (int sn = 0; sn < 2; sn++)
      for (int i = 0; i < 8; i++) {
        const int r = M0 + wm * 32 + sm * 16 + i + half * 8;
        const int o = N0 + wn * 32 + sn * 16 + l15;
        const float val = acc[sm][sn][i] + bias[o];
        const int t = r >> 4, b = r & 15;
        if (mode == 0) {
          const int h = o >> 7, e = o & 127;
          q_h[(((size_t)(b * 8 + h)) * SEQP + t) * DHEAD + e] = f2bf(val);
        } else if (mode == 1) {
          const int oo = o & 1023, h = oo >> 7, e = oo & 127;
          uint16_t* dst = (o < 1024) ? k_h : v_h;
          dst[(((size_t)(b * 8 + h)) * SEQP + t) * DHEAD + e] = f2bf(val);
        } else {
          if (t < 1056) {
            out_f32[(size_t)r * 1024 + o] = val;
          } else if (t < 1063) {
            const float c = fminf(fmaxf(val, -10.0f), 10.0f);
            out_f32[(size_t)OUT_MEMS_BASE + ((size_t)(t - 1056) * 16 + b) * 1024 + o] = c;
          }
        }
      }
}

// ---------------------------------------------------------------------------
// Flash attention: grid (9 q-tiles, 128 heads), 256 threads = 8 waves,
// each wave owns 16 q rows; K/V/mask tiles of 64 seq staged in LDS per block.
// ---------------------------------------------------------------------------
#define KLDS_STRIDE 136
#define VT_STRIDE   72
#define P_STRIDE    72

__global__ __launch_bounds__(256)
void attention_kernel(const uint16_t* __restrict__ q_h,
                      const uint16_t* __restrict__ k_h,
                      const uint16_t* __restrict__ v_h,
                      const unsigned char* __restrict__ amask, // [1064][1064]
                      const int* __restrict__ lengths,         // [16]
                      uint16_t* __restrict__ attn_bf) {        // [17024][1024]
  __shared__ __align__(16) uint16_t Klds[64 * KLDS_STRIDE];
  __shared__ __align__(16) uint16_t Vt[128 * VT_STRIDE];       // Vt[e][k]
  __shared__ __align__(16) uint16_t Plds[8][16 * P_STRIDE];
  __shared__ __align__(16) unsigned char Mlds[128 * 64];       // mask tile [q][k]

  const int tid  = threadIdx.x;
  const int lane = tid & 31;
  const int w    = tid >> 5;
  const int half = lane >> 4;
  const int l15  = lane & 15;
  const int nh   = blockIdx.y;        // head index (b*8 + h)
  const int b    = nh >> 3;
  const int h    = nh & 7;
  const int qtile = blockIdx.x * 128;
  const int qbase = qtile + w * 16;

  // klengths[b] = lengths[b] + M + (Q - max(lengths) - S)
  int maxlen = 0;
  for (int i = 0; i < 16; i++) maxlen = (lengths[i] > maxlen) ? lengths[i] : maxlen;
  const int klen = lengths[b] + 8 + (QTOT - maxlen - 8);

  // Q fragments for this wave's 16 rows: 4 chunks of K=32 over d=128
  Frag16 qfr[4];
  {
    const uint16_t* qrow = q_h + ((size_t)nh * SEQP + qbase + l15) * DHEAD;
    for (int kc = 0; kc < 4; kc++)
      for (int j = 0; j < 8; j++)
        qfr[kc].u[j] = *(const uint32_t*)(qrow + kc * 32 + a_koff(j, half));
  }

  float mrow[8], lrow[8];
  v8f accO[8];
  for (int i = 0; i < 8; i++) {
    mrow[i] = -3.0e38f; lrow[i] = 0.0f;
    for (int e = 0; e < 8; e++) accO[i][e] = 0.0f;
  }

  // staging coordinates (fixed per thread)
  const int srow = tid >> 2, sc0 = (tid & 3) * 32;   // K/V: 64 rows x 128 e
  const int mrow_q = tid >> 1, mc0 = (tid & 1) * 32; // mask: 128 q-rows x 64 k
  const int mqg = qtile + mrow_q;

  const int NKT = SEQP / 64;   // 18
  for (int kt = 0; kt < NKT; kt++) {
    const int kb = kt * 64;
    __syncthreads();
    // ---- stage K tile [64 seq][128 e] with b128 copies ----
    {
      const uint16_t* src = k_h + ((size_t)nh * SEQP + kb + srow) * DHEAD + sc0;
      uint16_t* dst = &Klds[srow * KLDS_STRIDE + sc0];
      for (int i = 0; i < 4; i++)
        *(uint4*)(dst + i * 8) = *(const uint4*)(src + i * 8);
    }
    // ---- stage V tile transposed: Vt[e][k] ----
    {
      const uint16_t* src = v_h + ((size_t)nh * SEQP + kb + srow) * DHEAD + sc0;
      for (int i = 0; i < 4; i++) {
        U4H8 u; u.q = *(const uint4*)(src + i * 8);
        for (int j = 0; j < 8; j++)
          Vt[(sc0 + i * 8 + j) * VT_STRIDE + srow] = u.h[j];
      }
    }
    // ---- stage mask tile [128 q][64 k]; OOB -> masked(1) ----
    {
      unsigned char* mdst = &Mlds[mrow_q * 64 + mc0];
      if (mqg >= QTOT) {
        for (int j = 0; j < 4; j++)
          *(uint2*)(mdst + j * 8) = make_uint2(0x01010101u, 0x01010101u);
      } else {
        const unsigned char* msrc = amask + (size_t)mqg * KTOT + kb + mc0;
        for (int j = 0; j < 4; j++) {
          const int kg0 = kb + mc0 + j * 8;
          if (kg0 + 7 < KTOT) {
            *(uint2*)(mdst + j * 8) = *(const uint2*)(msrc + j * 8);
          } else {
            for (int e = 0; e < 8; e++)
              mdst[j * 8 + e] = (kg0 + e < KTOT) ? msrc[j * 8 + e] : (unsigned char)1;
          }
        }
      }
    }
    // prefetch next tiles while computing this one
    if (kt + 1 < NKT) {
      const size_t nb = ((size_t)nh * SEQP + kb + 64 + srow) * DHEAD + sc0;
      __builtin_prefetch(k_h + nb, 0, 1);
      __builtin_prefetch(v_h + nb, 0, 1);
      if (mqg < QTOT && kb + 64 + mc0 < KTOT)
        __builtin_prefetch(amask + (size_t)mqg * KTOT + kb + 64 + mc0, 0, 1);
    }
    __syncthreads();

    // ---- scores: S(16q x 64k) = Q(16x128) * K^T ----
    v8f sc[4];
    for (int nt = 0; nt < 4; nt++) {
      v8f z; for (int e = 0; e < 8; e++) z[e] = 0.0f;
      for (int kc = 0; kc < 4; kc++) {
        Frag16 bf;
        const uint16_t* base = &Klds[(nt * 16 + l15) * KLDS_STRIDE + kc * 32];
        for (int j = 0; j < 8; j++)
          bf.u[j] = *(const uint32_t*)(base + b_koff(j, half));
        z = __builtin_amdgcn_wmma_f32_16x16x32_bf16(
            false, qfr[kc].v, false, bf.v, (short)0, z, false, false);
      }
      sc[nt] = z;
    }

    // ---- scale + masks (mask bytes from LDS) ----
    for (int nt = 0; nt < 4; nt++) {
      const int kg = kb + nt * 16 + l15;
      const bool kpad = (kg >= klen) || (kg >= KTOT);
      const unsigned char* mcol = &Mlds[(w * 16) * 64 + nt * 16 + l15];
      for (int i = 0; i < 8; i++) {
        const bool msk = kpad || (mcol[(i + half * 8) * 64] != 0);
        sc[nt][i] = msk ? NEG_INF_F : (sc[nt][i] * ATT_SCALE);
      }
    }

    // ---- online softmax (row = VGPR slot i + 8*half; cols across 16 lanes) --
    for (int i = 0; i < 8; i++) {
      float mt = sc[0][i];
      for (int nt = 1; nt < 4; nt++) mt = fmaxf(mt, sc[nt][i]);
      for (int d = 1; d < 16; d <<= 1) mt = fmaxf(mt, __shfl_xor(mt, d, 32));
      const float mn = fmaxf(mrow[i], mt);
      const float scale = __expf(mrow[i] - mn);
      float rsum = 0.0f;
      for (int nt = 0; nt < 4; nt++) {
        const float p = __expf(sc[nt][i] - mn);
        sc[nt][i] = p;
        rsum += p;
      }
      for (int d = 1; d < 16; d <<= 1) rsum += __shfl_xor(rsum, d, 32);
      lrow[i] = lrow[i] * scale + rsum;
      mrow[i] = mn;
      for (int j = 0; j < 8; j++) accO[j][i] *= scale;
    }

    // ---- P through per-wave LDS (C-layout -> A-layout transpose) ----
    uint16_t* P = Plds[w];
    for (int nt = 0; nt < 4; nt++)
      for (int i = 0; i < 8; i++)
        P[(i + half * 8) * P_STRIDE + nt * 16 + l15] = f2bf(sc[nt][i]);

    // ---- accO += P(16x64) * V(64x128) ----
    for (int kc = 0; kc < 2; kc++) {
      Frag16 pf;
      const uint16_t* pbase = &P[l15 * P_STRIDE + kc * 32];
      for (int j = 0; j < 8; j++)
        pf.u[j] = *(const uint32_t*)(pbase + a_koff(j, half));
      for (int jo = 0; jo < 8; jo++) {
        Frag16 vf;
        const uint16_t* vbase = &Vt[(jo * 16 + l15) * VT_STRIDE + kc * 32];
        for (int j = 0; j < 8; j++)
          vf.u[j] = *(const uint32_t*)(vbase + b_koff(j, half));
        accO[jo] = __builtin_amdgcn_wmma_f32_16x16x32_bf16(
            false, pf.v, false, vf.v, (short)0, accO[jo], false, false);
      }
    }
  }

  // ---- epilogue: attn[t*16+b][h*128 + e] = accO / l ----
  for (int i = 0; i < 8; i++) {
    const int t = qbase + i + half * 8;
    if (t >= QTOT) continue;
    const float inv = 1.0f / lrow[i];
    const size_t rbase = ((size_t)t * 16 + b) * 1024 + (size_t)h * 128;
    for (int jo = 0; jo < 8; jo++)
      attn_bf[rbase + jo * 16 + l15] = f2bf(accO[jo][i] * inv);
  }
}

// ---------------------------------------------------------------------------
// Host launcher
// ---------------------------------------------------------------------------
extern "C" void kernel_launch(void* const* d_in, const int* in_sizes, int n_in,
                              void* d_out, int out_size, void* d_ws, size_t ws_size,
                              hipStream_t stream) {
  const float* utter = (const float*)d_in[0];
  const int*   lengths = (const int*)d_in[1];
  const float* rctx  = (const float*)d_in[2];
  const float* summ  = (const float*)d_in[3];
  const float* memsp = (const float*)d_in[4];
  const unsigned char* amask = (const unsigned char*)d_in[5];
  const float* Wq  = (const float*)d_in[6];
  const float* bq  = (const float*)d_in[7];
  const float* Wkv = (const float*)d_in[8];
  const float* bkv = (const float*)d_in[9];
  const float* Wo  = (const float*)d_in[10];
  const float* bo  = (const float*)d_in[11];

  char* ws = (char*)d_ws;
  uint16_t* wq_bf  = (uint16_t*)ws; ws += (size_t)1024 * 1024 * 2;
  uint16_t* wkv_bf = (uint16_t*)ws; ws += (size_t)2048 * 1024 * 2;
  uint16_t* wo_bf  = (uint16_t*)ws; ws += (size_t)1024 * 1024 * 2;
  uint16_t* q_h    = (uint16_t*)ws; ws += (size_t)NHEADS * SEQP * DHEAD * 2;
  uint16_t* k_h    = (uint16_t*)ws; ws += (size_t)NHEADS * SEQP * DHEAD * 2;
  uint16_t* v_h    = (uint16_t*)ws; ws += (size_t)NHEADS * SEQP * DHEAD * 2;
  uint16_t* attn   = (uint16_t*)ws;

  // weights -> bf16
  cvt_f32_bf16_kernel<<<dim3(1024), dim3(256), 0, stream>>>(Wq,  wq_bf,  1024 * 1024);
  cvt_f32_bf16_kernel<<<dim3(2048), dim3(256), 0, stream>>>(Wkv, wkv_bf, 2048 * 1024);
  cvt_f32_bf16_kernel<<<dim3(1024), dim3(256), 0, stream>>>(Wo,  wo_bf,  1024 * 1024);

  // zero padded q/k/v head buffers (contiguous in workspace)
  const int nzero = (int)((size_t)3 * NHEADS * SEQP * DHEAD * 2 / 4);
  zero_u32_kernel<<<dim3(4096), dim3(256), 0, stream>>>((uint32_t*)q_h, nzero);

  // Q projection -> q_heads bf16
  proj_gemm_kernel<<<dim3(266, 8), 256, 0, stream>>>(
      0, utter, rctx, summ, memsp, nullptr, wq_bf, bq, q_h, k_h, v_h, nullptr);
  // KV projection -> k_heads/v_heads bf16
  proj_gemm_kernel<<<dim3(266, 16), 256, 0, stream>>>(
      1, utter, rctx, summ, memsp, nullptr, wkv_bf, bkv, q_h, k_h, v_h, nullptr);

  // flash attention -> attn bf16 [17024 x 1024]
  attention_kernel<<<dim3(9, 128), 256, 0, stream>>>(
      q_h, k_h, v_h, amask, lengths, attn);

  // output projection + bias + clip + output/mems scatter -> d_out f32
  proj_gemm_kernel<<<dim3(266, 8), 256, 0, stream>>>(
      2, utter, rctx, summ, memsp, attn, wo_bf, bo, q_h, k_h, v_h, (float*)d_out);
}
